// Net_84696755077754
// MI455X (gfx1250) — matrix operand order repaired and checked by reading, not compiled, not run on hardware
//
#include <hip/hip_runtime.h>
#include <hip/hip_bf16.h>

typedef _Float16 f16;
typedef _Float16 v16h __attribute__((ext_vector_type(16)));
typedef _Float16 v8h  __attribute__((ext_vector_type(8)));
typedef float    v8f  __attribute__((ext_vector_type(8)));

// ---------------- workspace layout (f16 elements) ----------------
// All weight matrices stored row-major [N][Kpad], zero padded, f16.
#define W1_N 448   // conv1 dense: N = 3ch*144px -> pad 448
#define W1_K 160   // K = 144 grid cells -> pad 160 (5 k-tiles of 32)
#define W2_N 224   // conv2 dense: N = 6ch*36px -> pad 224
#define W2_K 128   // K = 3ch*36px = 108 -> pad 128
#define W3_N 128   // fc1: N = 128
#define W3_K 64    // K = 54 -> pad 64
#define W4_N 128   // out: N = 128
#define W4_K 128   // K = 128
#define W1_OFF 0
#define W2_OFF (W1_OFF + W1_N * W1_K)          // 71680
#define W3_OFF (W2_OFF + W2_N * W2_K)          // 100352
#define W4_OFF (W3_OFF + W3_N * W3_K)          // 108544
#define WS_ELEMS (W4_OFF + W4_N * W4_K)        // 124928 f16 = 244 KB

// ---------------- LDS layout (f16 elements), 64 samples / WG -----
#define SPW 64
#define A1_OFF 0                                // [64][160] x grid, f16
#define C_OFF  (A1_OFF + SPW * W1_K)            // [64][448] gemm scratch
#define A2_OFF (C_OFF  + SPW * W1_N)            // [64][128] h1 pooled
#define A3_OFF (A2_OFF + SPW * W2_K)            // [64][64]  h2 pooled
#define A4_OFF (A3_OFF + SPW * W3_K)            // [64][128] fc1 act
#define SMEM_ELEMS (A4_OFF + SPW * W4_K)        // 59392 f16 = 118784 B

// =================================================================
// Prep kernel: densify conv weights + pad/convert FC weights to f16
// =================================================================
__global__ __launch_bounds__(256) void net_prep(
    const float* __restrict__ w1,   // (3,1,3,3)
    const float* __restrict__ w2,   // (6,3,3,3)
    const float* __restrict__ w3,   // (128,54)
    const float* __restrict__ w4,   // (128,128)
    f16* __restrict__ ws) {
  int idx = blockIdx.x * 256 + threadIdx.x;
  if (idx >= WS_ELEMS) return;
  float val = 0.0f;
  if (idx < W2_OFF) {                           // W1d [448][160]
    int n = idx / W1_K, k = idx % W1_K;
    if (n < 432 && k < 144) {
      int c = n / 144, p = n % 144;
      int oy = p / 12, ox = p % 12;
      int iy = k / 12, ix = k % 12;
      int dy = iy - oy + 1, dx = ix - ox + 1;
      if (dy >= 0 && dy < 3 && dx >= 0 && dx < 3)
        val = w1[c * 9 + dy * 3 + dx];
    }
  } else if (idx < W3_OFF) {                    // W2d [224][128]
    int j = idx - W2_OFF;
    int n = j / W2_K, k = j % W2_K;
    if (n < 216 && k < 108) {
      int c = n / 36, p = n % 36;
      int oy = p / 6, ox = p % 6;
      int ci = k / 36, q = k % 36;
      int iy = q / 6, ix = q % 6;
      int dy = iy - oy + 1, dx = ix - ox + 1;
      if (dy >= 0 && dy < 3 && dx >= 0 && dx < 3)
        val = w2[((c * 3 + ci) * 3 + dy) * 3 + dx];
    }
  } else if (idx < W4_OFF) {                    // W3 [128][64]
    int j = idx - W3_OFF;
    int n = j / W3_K, k = j % W3_K;
    if (k < 54) val = w3[n * 54 + k];
  } else {                                      // W4 [128][128]
    val = w4[idx - W4_OFF];
  }
  ws[idx] = (f16)val;
}

// =================================================================
// Fragment loaders per CDNA5 ISA WMMA layouts (wave32)
// =================================================================
// A (16x32 f16, row-major src [.. ][Kpad]): lanes 0-15 -> K 0..7 & 16..23,
// lanes 16-31 -> K 8..15 & 24..31, M = lane&15.
__device__ __forceinline__ v16h load_a_frag(const f16* A, int Kpad, int mt,
                                            int kt, int lane) {
  const f16* p = A + (mt * 16 + (lane & 15)) * Kpad + kt * 32 + ((lane >> 4) * 8);
  v8h lo = *(const v8h*)p;
  v8h hi = *(const v8h*)(p + 16);
  return __builtin_shufflevector(lo, hi, 0, 1, 2, 3, 4, 5, 6, 7,
                                 8, 9, 10, 11, 12, 13, 14, 15);
}
// B (32x16 f16 from row-major W[N][Kpad]): N = lane&15,
// lanes 0-15 -> K 0..7 & 16..23, lanes 16-31 -> K 8..15 & 24..31.
__device__ __forceinline__ v16h load_b_frag(const f16* W, int Kpad, int nt,
                                            int kt, int lane) {
  const f16* p = W + (nt * 16 + (lane & 15)) * Kpad + kt * 32 + ((lane >> 4) * 8);
  v8h lo = *(const v8h*)p;
  v8h hi = *(const v8h*)(p + 16);
  return __builtin_shufflevector(lo, hi, 0, 1, 2, 3, 4, 5, 6, 7,
                                 8, 9, 10, 11, 12, 13, 14, 15);
}

#define ZERO8F {0.f, 0.f, 0.f, 0.f, 0.f, 0.f, 0.f, 0.f}

// =================================================================
// Fused network kernel: 4 WMMA GEMM stages + pool/relu epilogues
// =================================================================
__global__ __launch_bounds__(256) void net_fused(
    const float* __restrict__ x,
    const float* __restrict__ b1, const float* __restrict__ b2,
    const float* __restrict__ b3, const float* __restrict__ b4,
    const f16* __restrict__ ws, float* __restrict__ out, int num) {
  extern __shared__ __align__(16) f16 smem[];
  f16* A1 = smem + A1_OFF;
  f16* Cb = smem + C_OFF;
  f16* A2 = smem + A2_OFF;
  f16* A3 = smem + A3_OFF;
  f16* A4 = smem + A4_OFF;

  const int tid  = threadIdx.x;
  const int lane = tid & 31;
  const int wv   = tid >> 5;           // 8 waves per WG
  const int s0   = blockIdx.x * SPW;

  // keep the (tiny, L2-resident) weight matrices warm
  __builtin_prefetch(ws + (tid & 255) * 488, 0, 1);

  // ---- stage 0: load x tile, zero-pad 12x12 grid tail, f32 -> f16 ----
  for (int i = tid; i < SPW * W1_K; i += 256) {
    int s = i / W1_K, k = i % W1_K;
    float v = 0.0f;
    if (k < 128 && (s0 + s) < num) v = x[(size_t)(s0 + s) * 128 + k];
    A1[i] = (f16)v;
  }
  __syncthreads();

  // ---- GEMM1: C[64][448] = A1[64][160] @ W1d^T, + conv1_b ----
  for (int t = wv; t < 4 * (W1_N / 16); t += 8) {
    int mt = t / (W1_N / 16), nt = t % (W1_N / 16);
    v8f acc = ZERO8F;
#pragma unroll
    for (int kt = 0; kt < W1_K / 32; ++kt) {
      v16h a = load_a_frag(A1, W1_K, mt, kt, lane);
      v16h b = load_b_frag(ws + W1_OFF, W1_K, nt, kt, lane);
      acc = __builtin_amdgcn_wmma_f32_16x16x32_f16(false, a, false, b,
                                                   (short)0, acc, false, false);
    }
    int col = nt * 16 + (lane & 15);
    float bias = (col < 432) ? b1[col / 144] : 0.0f;
    int rb = mt * 16 + ((lane >> 4) * 8);
#pragma unroll
    for (int r = 0; r < 8; ++r)
      Cb[(rb + r) * W1_N + col] = (f16)(acc[r] + bias);
  }
  __syncthreads();

  // ---- pool1: 2x2 max over 12x12 per channel, relu -> A2[64][128] ----
  for (int i = tid; i < SPW * W2_K; i += 256) {
    int s = i >> 7, j = i & 127;
    float v = 0.0f;
    if (j < 108) {
      int c = j / 36, q = j % 36, py = q / 6, px = q % 6;
      const f16* r0 = &Cb[s * W1_N + c * 144 + (py * 2) * 12 + px * 2];
      float m = fmaxf(fmaxf((float)r0[0], (float)r0[1]),
                      fmaxf((float)r0[12], (float)r0[13]));
      v = fmaxf(m, 0.0f);
    }
    A2[i] = (f16)v;
  }
  __syncthreads();

  // ---- GEMM2: C[64][224] = A2[64][128] @ W2d^T, + conv2_b ----
  for (int t = wv; t < 4 * (W2_N / 16); t += 8) {
    int mt = t / (W2_N / 16), nt = t % (W2_N / 16);
    v8f acc = ZERO8F;
#pragma unroll
    for (int kt = 0; kt < W2_K / 32; ++kt) {
      v16h a = load_a_frag(A2, W2_K, mt, kt, lane);
      v16h b = load_b_frag(ws + W2_OFF, W2_K, nt, kt, lane);
      acc = __builtin_amdgcn_wmma_f32_16x16x32_f16(false, a, false, b,
                                                   (short)0, acc, false, false);
    }
    int col = nt * 16 + (lane & 15);
    float bias = (col < 216) ? b2[col / 36] : 0.0f;
    int rb = mt * 16 + ((lane >> 4) * 8);
#pragma unroll
    for (int r = 0; r < 8; ++r)
      Cb[(rb + r) * W2_N + col] = (f16)(acc[r] + bias);
  }
  __syncthreads();

  // ---- pool2: 2x2 max over 6x6 per channel, relu -> A3[64][64] ----
  for (int i = tid; i < SPW * W3_K; i += 256) {
    int s = i >> 6, j = i & 63;
    float v = 0.0f;
    if (j < 54) {
      int c = j / 9, q = j % 9, py = q / 3, px = q % 3;
      const f16* r0 = &Cb[s * W2_N + c * 36 + (py * 2) * 6 + px * 2];
      float m = fmaxf(fmaxf((float)r0[0], (float)r0[1]),
                      fmaxf((float)r0[6], (float)r0[7]));
      v = fmaxf(m, 0.0f);
    }
    A3[i] = (f16)v;
  }
  __syncthreads();

  // ---- GEMM3 (fc1): A4[64][128] = relu(A3[64][64] @ W3^T + b3) ----
  for (int t = wv; t < 4 * (W3_N / 16); t += 8) {
    int mt = t / (W3_N / 16), nt = t % (W3_N / 16);
    v8f acc = ZERO8F;
#pragma unroll
    for (int kt = 0; kt < W3_K / 32; ++kt) {
      v16h a = load_a_frag(A3, W3_K, mt, kt, lane);
      v16h b = load_b_frag(ws + W3_OFF, W3_K, nt, kt, lane);
      acc = __builtin_amdgcn_wmma_f32_16x16x32_f16(false, a, false, b,
                                                   (short)0, acc, false, false);
    }
    int col = nt * 16 + (lane & 15);
    float bias = b3[col];
    int rb = mt * 16 + ((lane >> 4) * 8);
#pragma unroll
    for (int r = 0; r < 8; ++r)
      A4[(rb + r) * W4_K + col] = (f16)fmaxf(acc[r] + bias, 0.0f);
  }
  __syncthreads();

  // ---- GEMM4 (out): d_out = A4[64][128] @ W4^T + b4 (f32 to HBM) ----
  for (int t = wv; t < 4 * (W4_N / 16); t += 8) {
    int mt = t / (W4_N / 16), nt = t % (W4_N / 16);
    v8f acc = ZERO8F;
#pragma unroll
    for (int kt = 0; kt < W4_K / 32; ++kt) {
      v16h a = load_a_frag(A4, W4_K, mt, kt, lane);
      v16h b = load_b_frag(ws + W4_OFF, W4_K, nt, kt, lane);
      acc = __builtin_amdgcn_wmma_f32_16x16x32_f16(false, a, false, b,
                                                   (short)0, acc, false, false);
    }
    int col = nt * 16 + (lane & 15);
    float bias = b4[col];
    int rb = mt * 16 + ((lane >> 4) * 8);
#pragma unroll
    for (int r = 0; r < 8; ++r) {
      int row = s0 + rb + r;
      if (row < num) out[(size_t)row * 128 + col] = acc[r] + bias;
    }
  }
}

// =================================================================
extern "C" void kernel_launch(void* const* d_in, const int* in_sizes, int n_in,
                              void* d_out, int out_size, void* d_ws,
                              size_t ws_size, hipStream_t stream) {
  const float* x  = (const float*)d_in[0];
  const float* w1 = (const float*)d_in[1];
  const float* b1 = (const float*)d_in[2];
  const float* w2 = (const float*)d_in[3];
  const float* b2 = (const float*)d_in[4];
  const float* w3 = (const float*)d_in[5];
  const float* b3 = (const float*)d_in[6];
  const float* w4 = (const float*)d_in[7];
  const float* b4 = (const float*)d_in[8];
  f16* ws = (f16*)d_ws;
  int num = in_sizes[0] / 128;

  net_prep<<<(WS_ELEMS + 255) / 256, 256, 0, stream>>>(w1, w2, w3, w4, ws);
  net_fused<<<(num + SPW - 1) / SPW, 256, SMEM_ELEMS * sizeof(f16), stream>>>(
      x, b1, b2, b3, b4, ws, (float*)d_out, num);
}
